// InteractionGNNBlock_57251914056453
// MI455X (gfx1250) — matrix operand order repaired and checked by reading, not compiled, not run on hardware
//
#include <hip/hip_runtime.h>
#include <hip/hip_bf16.h>

// ---------------------------------------------------------------------------
// InteractionGNNBlock for MI455X (gfx1250, wave32, WMMA).
// GEMMs in bf16 via v_wmma_f32_16x16x32_bf16 (fp32 accumulate);
// LayerNorm / residuals / activations in fp32.
// M=64 rows/block, B-fragment reused across 4 M-tiles (4 accumulators/wave).
// ---------------------------------------------------------------------------

typedef __bf16 bf16;
typedef __attribute__((ext_vector_type(16))) __bf16 v16bf;
typedef __attribute__((ext_vector_type(8)))  __bf16 v8bf;
typedef __attribute__((ext_vector_type(8)))  float  v8f;

#define NNODES 50000
#define NEDGES 600000
#define LAT    128
#define LNEPS  1e-5f
#define MROWS  64              // rows per block (4 x 16 M-tiles)

// ---------------------------------------------------------------------------
// Linear layer: sY[MROWS][Nf] (fp32) = sA[MROWS][K] (bf16) @ W + bias.
// 8 waves split the Nf/16 output tiles; each wave loads one B fragment per
// K-step and reuses it across the 4 M-tiles (4 fp32 accumulators).
// A-fragment lane layout (contiguous k order used here, same as B; 16x32):
//   lanes 0-15:  M=lane,    K = kt..kt+15
//   lanes 16-31: M=lane-16, K = kt+16..kt+31
// B-fragment: col = lane&15, 16 contiguous K values at kt + (lane<16?0:16)
//   (weights stored transposed [Nf][K] bf16 so this is one 32B load).
// C/D mapping: VGPR r -> row r + 8*(lane>=16), col = lane&15.
// ---------------------------------------------------------------------------
__device__ __forceinline__ void gemm_layer(const bf16* __restrict__ sA,
                                           float* __restrict__ sY,
                                           const bf16* __restrict__ wt,
                                           const float* __restrict__ bias,
                                           int K, int Nf, int tid) {
  const int wave = tid >> 5;
  const int lane = tid & 31;
  const int col  = lane & 15;
  const int crb  = (lane < 16) ? 0 : 8;
  const int arow = lane & 15;
  const int akb  = (lane < 16) ? 0 : 16;
  const int bkb  = (lane < 16) ? 0 : 16;
  for (int nt = wave; nt < (Nf >> 4); nt += 8) {
    v8f acc0 = {}, acc1 = {}, acc2 = {}, acc3 = {};
    const bf16* __restrict__ brow = wt + (size_t)(nt * 16 + col) * K;
    for (int kt = 0; kt < K; kt += 32) {
      v16bf b = *(const v16bf*)(brow + kt + bkb);
#pragma unroll
      for (int m = 0; m < 4; ++m) {
        const bf16* __restrict__ ap = sA + (m * 16 + arow) * K + kt + akb;
        v8bf alo = *(const v8bf*)(ap);
        v8bf ahi = *(const v8bf*)(ap + 8);
        v16bf a  = __builtin_shufflevector(alo, ahi,
                     0, 1, 2, 3, 4, 5, 6, 7, 8, 9, 10, 11, 12, 13, 14, 15);
        v8f cin = (m == 0) ? acc0 : (m == 1) ? acc1 : (m == 2) ? acc2 : acc3;
        v8f cout = __builtin_amdgcn_wmma_f32_16x16x32_bf16(
                       false, a, false, b, (short)0, cin, false, false);
        if      (m == 0) acc0 = cout;
        else if (m == 1) acc1 = cout;
        else if (m == 2) acc2 = cout;
        else             acc3 = cout;
      }
    }
    const float bs = bias[nt * 16 + col];
#pragma unroll
    for (int m = 0; m < 4; ++m) {
      v8f acc = (m == 0) ? acc0 : (m == 1) ? acc1 : (m == 2) ? acc2 : acc3;
#pragma unroll
      for (int r = 0; r < 8; ++r)
        sY[(m * 16 + crb + r) * Nf + nt * 16 + col] = acc[r] + bs;
    }
  }
}

// Row mean / rstd over Nf features; 16 threads per row (one half-wave each),
// reduced with shfl_xor width 16 (wave32-safe).
__device__ __forceinline__ void ln_stats(const float* __restrict__ sY, int Nf,
                                         int row, int j, float& mean, float& rstd) {
  float s = 0.f, ss = 0.f;
  for (int c = j; c < Nf; c += 16) {
    float v = sY[row * Nf + c];
    s += v; ss += v * v;
  }
#pragma unroll
  for (int o = 8; o >= 1; o >>= 1) {
    s  += __shfl_xor(s,  o, 16);
    ss += __shfl_xor(ss, o, 16);
  }
  const float inv = 1.0f / (float)Nf;
  mean = s * inv;
  const float var = ss * inv - mean * mean;
  rstd = rsqrtf(var + LNEPS);
}

// ---------------------------------------------------------------------------
// Stage 0: utility kernels
// ---------------------------------------------------------------------------
__global__ __launch_bounds__(256)
void zero_f32_kernel(float* __restrict__ p, long n) {
  long i = (long)blockIdx.x * 256 + threadIdx.x;
  if (i < n) p[i] = 0.f;
}

__global__ __launch_bounds__(256)
void scatter_add_kernel(const float* __restrict__ edges,
                        const int* __restrict__ dst,
                        float* __restrict__ msg) {
  long i = (long)blockIdx.x * 256 + threadIdx.x;
  if (i < (long)NEDGES * LAT) {
    long e = i >> 7;              // /128
    int  c = (int)(i & 127);
    atomicAdd(&msg[(long)dst[e] * LAT + c], edges[i]);
  }
}

// fp32 [K][Nf]  ->  bf16 transposed [Nf][K]
__global__ __launch_bounds__(256)
void transpose_w_kernel(const float* __restrict__ w, bf16* __restrict__ wt,
                        int K, int Nf) {
  long i = (long)blockIdx.x * 256 + threadIdx.x;
  if (i < (long)K * Nf) {
    int k = (int)(i / Nf);
    int n = (int)(i % Nf);
    wt[(long)n * K + k] = (bf16)w[i];
  }
}

// ---------------------------------------------------------------------------
// Stage 1: fused node MLP.  MROWS nodes / block, 256 threads (8 waves).
//   X0 = concat(nodes, msg) [M x 256] -> (Lin+LN+ReLU)x2 -> Lin+LN+ReLU
//   out = act + nodes (residual); also store bf16 copy for edge gathers.
// Rows past NNODES are clamped on load (EXEC stays all-ones for WMMA) and
// guarded on store.
// ---------------------------------------------------------------------------
__global__ __launch_bounds__(256)
void node_mlp_kernel(const float* __restrict__ nodes, const float* __restrict__ msg,
                     const bf16* __restrict__ w0t, const bf16* __restrict__ w1t,
                     const bf16* __restrict__ w2t,
                     const float* __restrict__ b0, const float* __restrict__ b1,
                     const float* __restrict__ b2,
                     const float* __restrict__ g0, const float* __restrict__ g1,
                     const float* __restrict__ g2,
                     const float* __restrict__ be0, const float* __restrict__ be1,
                     const float* __restrict__ be2,
                     float* __restrict__ out_nodes, bf16* __restrict__ nodesb) {
  __shared__ bf16  sA[MROWS * 256];
  __shared__ float sY[MROWS * 256];
  const int  tid  = threadIdx.x;
  const long row0 = (long)blockIdx.x * MROWS;

  // Load X0 = [nodes | msg] as bf16 (clamped rows keep EXEC uniform)
  for (int idx = tid; idx < MROWS * 256; idx += 256) {
    int  r  = idx >> 8;
    int  c  = idx & 255;
    long gr = row0 + r;
    if (gr >= NNODES) gr = NNODES - 1;
    float v = (c < LAT) ? nodes[gr * LAT + c] : msg[gr * LAT + (c - LAT)];
    sA[r * 256 + c] = (bf16)v;
  }
  __syncthreads();

  // ---- layer 0: 256 -> 256, LN + ReLU
  gemm_layer(sA, sY, w0t, b0, 256, 256, tid);
  __syncthreads();
  {
    int j = tid & 15;
    for (int row = tid >> 4; row < MROWS; row += 16) {
      float m, rs; ln_stats(sY, 256, row, j, m, rs);
      for (int c = j; c < 256; c += 16) {
        float v = (sY[row * 256 + c] - m) * rs * g0[c] + be0[c];
        sA[row * 256 + c] = (bf16)fmaxf(v, 0.f);
      }
    }
  }
  __syncthreads();

  // ---- layer 1: 256 -> 256, LN + ReLU
  gemm_layer(sA, sY, w1t, b1, 256, 256, tid);
  __syncthreads();
  {
    int j = tid & 15;
    for (int row = tid >> 4; row < MROWS; row += 16) {
      float m, rs; ln_stats(sY, 256, row, j, m, rs);
      for (int c = j; c < 256; c += 16) {
        float v = (sY[row * 256 + c] - m) * rs * g1[c] + be1[c];
        sA[row * 256 + c] = (bf16)fmaxf(v, 0.f);
      }
    }
  }
  __syncthreads();

  // ---- layer 2: 256 -> 128, LN + ReLU, residual, dual write
  gemm_layer(sA, sY, w2t, b2, 256, 128, tid);
  __syncthreads();
  {
    int j = tid & 15;
    for (int row = tid >> 4; row < MROWS; row += 16) {
      float m, rs; ln_stats(sY, 128, row, j, m, rs);
      long gr = row0 + row;
      if (gr < NNODES) {
        for (int c = j; c < 128; c += 16) {
          float v = (sY[row * 128 + c] - m) * rs * g2[c] + be2[c];
          v = fmaxf(v, 0.f);
          float o = v + nodes[gr * LAT + c];
          out_nodes[gr * LAT + c] = o;
          nodesb[gr * LAT + c]    = (bf16)o;
        }
      }
    }
  }
}

// ---------------------------------------------------------------------------
// Stage 2: fused edge MLP.  MROWS edges / block (E % 64 == 0, no guards).
//   X0 = concat(nn[src], nn[dst], edges) [M x 384] (nn gathered from bf16
//   table resident in L2) -> (Lin+LN+ReLU)x2 -> Lin+LN+tanh ; + residual.
// ---------------------------------------------------------------------------
__global__ __launch_bounds__(256)
void edge_mlp_kernel(const float* __restrict__ edges, const bf16* __restrict__ nodesb,
                     const int* __restrict__ graph,
                     const bf16* __restrict__ w0t, const bf16* __restrict__ w1t,
                     const bf16* __restrict__ w2t,
                     const float* __restrict__ b0, const float* __restrict__ b1,
                     const float* __restrict__ b2,
                     const float* __restrict__ g0, const float* __restrict__ g1,
                     const float* __restrict__ g2,
                     const float* __restrict__ be0, const float* __restrict__ be1,
                     const float* __restrict__ be2,
                     float* __restrict__ out_edges) {
  __shared__ bf16  sA[MROWS * 384];
  __shared__ float sY[MROWS * 256];
  __shared__ int   sSrc[MROWS], sDst[MROWS];
  const int  tid = threadIdx.x;
  const long eb  = (long)blockIdx.x * MROWS;

  if (tid < MROWS)            sSrc[tid]         = graph[eb + tid];
  else if (tid < 2 * MROWS)   sDst[tid - MROWS] = graph[(long)NEDGES + eb + (tid - MROWS)];
  __syncthreads();

  // Gather X0 = [nn[src] | nn[dst] | edge] as bf16
  for (int idx = tid; idx < MROWS * 384; idx += 256) {
    int r = idx / 384;
    int c = idx % 384;
    bf16 v;
    if (c < 128)      v = nodesb[(long)sSrc[r] * LAT + c];
    else if (c < 256) v = nodesb[(long)sDst[r] * LAT + (c - 128)];
    else              v = (bf16)edges[(eb + r) * LAT + (c - 256)];
    sA[r * 384 + c] = v;
  }
  __syncthreads();

  // ---- layer 0: 384 -> 256, LN + ReLU
  gemm_layer(sA, sY, w0t, b0, 384, 256, tid);
  __syncthreads();
  {
    int j = tid & 15;
    for (int row = tid >> 4; row < MROWS; row += 16) {
      float m, rs; ln_stats(sY, 256, row, j, m, rs);
      for (int c = j; c < 256; c += 16) {
        float v = (sY[row * 256 + c] - m) * rs * g0[c] + be0[c];
        sA[row * 256 + c] = (bf16)fmaxf(v, 0.f);   // stride becomes 256
      }
    }
  }
  __syncthreads();

  // ---- layer 1: 256 -> 256, LN + ReLU
  gemm_layer(sA, sY, w1t, b1, 256, 256, tid);
  __syncthreads();
  {
    int j = tid & 15;
    for (int row = tid >> 4; row < MROWS; row += 16) {
      float m, rs; ln_stats(sY, 256, row, j, m, rs);
      for (int c = j; c < 256; c += 16) {
        float v = (sY[row * 256 + c] - m) * rs * g1[c] + be1[c];
        sA[row * 256 + c] = (bf16)fmaxf(v, 0.f);
      }
    }
  }
  __syncthreads();

  // ---- layer 2: 256 -> 128, LN + tanh, residual
  gemm_layer(sA, sY, w2t, b2, 256, 128, tid);
  __syncthreads();
  {
    int j = tid & 15;
    for (int row = tid >> 4; row < MROWS; row += 16) {
      float m, rs; ln_stats(sY, 128, row, j, m, rs);
      long ge = eb + row;
      for (int c = j; c < 128; c += 16) {
        float v = (sY[row * 128 + c] - m) * rs * g2[c] + be2[c];
        out_edges[ge * LAT + c] = tanhf(v) + edges[ge * LAT + c];
      }
    }
  }
}

// ---------------------------------------------------------------------------
// Host-side orchestration
// ---------------------------------------------------------------------------
extern "C" void kernel_launch(void* const* d_in, const int* in_sizes, int n_in,
                              void* d_out, int out_size, void* d_ws, size_t ws_size,
                              hipStream_t stream) {
  (void)in_sizes; (void)n_in; (void)out_size; (void)ws_size;

  const float* nodes = (const float*)d_in[0];
  const float* edges = (const float*)d_in[1];
  const int*   graph = (const int*)d_in[2];
  // node_params: {"W":[3], "b":[3], "g":[3], "beta":[3]} flattened in order
  const float* nW[3]  = {(const float*)d_in[3],  (const float*)d_in[4],  (const float*)d_in[5]};
  const float* nB[3]  = {(const float*)d_in[6],  (const float*)d_in[7],  (const float*)d_in[8]};
  const float* nG[3]  = {(const float*)d_in[9],  (const float*)d_in[10], (const float*)d_in[11]};
  const float* nBe[3] = {(const float*)d_in[12], (const float*)d_in[13], (const float*)d_in[14]};
  const float* eW[3]  = {(const float*)d_in[15], (const float*)d_in[16], (const float*)d_in[17]};
  const float* eB[3]  = {(const float*)d_in[18], (const float*)d_in[19], (const float*)d_in[20]};
  const float* eG[3]  = {(const float*)d_in[21], (const float*)d_in[22], (const float*)d_in[23]};
  const float* eBe[3] = {(const float*)d_in[24], (const float*)d_in[25], (const float*)d_in[26]};

  float* out_nodes = (float*)d_out;
  float* out_edges = (float*)d_out + (size_t)NNODES * LAT;

  // Workspace layout
  char*  ws     = (char*)d_ws;
  size_t off    = 0;
  float* msg    = (float*)(ws + off); off += (size_t)NNODES * LAT * sizeof(float);
  bf16*  nodesb = (bf16*)(ws + off);  off += (size_t)NNODES * LAT * sizeof(bf16);
  off = (off + 255) & ~(size_t)255;
  bf16* nWt[3]; bf16* eWt[3];
  const int nK[3] = {256, 256, 256}, nN[3] = {256, 256, 128};   // node W shapes [K][N]
  const int eK[3] = {384, 256, 256}, eN[3] = {256, 256, 128};   // edge W shapes [K][N]
  for (int i = 0; i < 3; ++i) { nWt[i] = (bf16*)(ws + off); off += (size_t)nK[i] * nN[i] * sizeof(bf16); }
  for (int i = 0; i < 3; ++i) { eWt[i] = (bf16*)(ws + off); off += (size_t)eK[i] * eN[i] * sizeof(bf16); }

  // 0) zero edge-message accumulator
  {
    long n = (long)NNODES * LAT;
    zero_f32_kernel<<<(unsigned)((n + 255) / 256), 256, 0, stream>>>(msg, n);
  }
  // 1) weight transpose + bf16 convert (tiny, L2-resident afterwards)
  for (int i = 0; i < 3; ++i) {
    long n = (long)nK[i] * nN[i];
    transpose_w_kernel<<<(unsigned)((n + 255) / 256), 256, 0, stream>>>(nW[i], nWt[i], nK[i], nN[i]);
  }
  for (int i = 0; i < 3; ++i) {
    long n = (long)eK[i] * eN[i];
    transpose_w_kernel<<<(unsigned)((n + 255) / 256), 256, 0, stream>>>(eW[i], eWt[i], eK[i], eN[i]);
  }
  // 2) scatter-add edges -> messages
  {
    long n = (long)NEDGES * LAT;
    scatter_add_kernel<<<(unsigned)((n + 255) / 256), 256, 0, stream>>>(edges, graph + NEDGES, msg);
  }
  // 3) fused node MLP (+residual), emits fp32 new_nodes and bf16 gather table
  node_mlp_kernel<<<(NNODES + MROWS - 1) / MROWS, 256, 0, stream>>>(
      nodes, msg, nWt[0], nWt[1], nWt[2],
      nB[0], nB[1], nB[2], nG[0], nG[1], nG[2], nBe[0], nBe[1], nBe[2],
      out_nodes, nodesb);
  // 4) fused edge MLP (+residual)
  edge_mlp_kernel<<<NEDGES / MROWS, 256, 0, stream>>>(
      edges, nodesb, graph, eWt[0], eWt[1], eWt[2],
      eB[0], eB[1], eB[2], eG[0], eG[1], eG[2], eBe[0], eBe[1], eBe[2],
      out_edges);
}
